// GrassmannMixingBlock_5239860101722
// MI455X (gfx1250) — matrix operand order repaired and checked by reading, not compiled, not run on hardware
//
#include <hip/hip_runtime.h>
#include <math.h>

typedef __attribute__((ext_vector_type(2)))  float  v2f;
typedef __attribute__((ext_vector_type(8)))  float  v8f;
typedef __attribute__((ext_vector_type(16))) __bf16 v16bf;
typedef unsigned short ushort_t;
typedef unsigned int   uint_t;

#define D_MODEL 1024
#define BATCH   4
#define SEQ     4096
#define NTOK    (BATCH * SEQ)      // 16384
#define RDIM    32
#define PDIM    496
#define PSTRIDE 512                // padded plucker-sum K (zeros in pad)
#define KGATE   2048

// ---- workspace layout ----
// float region (in floats)
#define Z_OFF     ((size_t)0)
#define SBAR_OFF  ((size_t)NTOK * RDIM)                    // 524288
#define G_OFF     (SBAR_OFF + (size_t)NTOK * PSTRIDE)      // 8912896
#define STATS_OFF (G_OFF + (size_t)NTOK * D_MODEL)         // 25690112 (NTOK*4 floats)
#define FLOAT_END (STATS_OFF + (size_t)NTOK * 4)           // 25755648
// ushort (bf16) region, offsets in ushorts, based at ws_u = (ushort*)(ws + FLOAT_END)
#define WGT_HI_OFF ((size_t)0)                              // [n][k] n<1024,k<2048
#define WGT_LO_OFF ((size_t)D_MODEL * KGATE)
#define WPT_HI_OFF (WGT_LO_OFF + (size_t)D_MODEL * KGATE)   // [n][k] k<512
#define WPT_LO_OFF (WPT_HI_OFF + (size_t)D_MODEL * PSTRIDE)

union FragBF { v16bf v; uint4 q[2]; };

__device__ __forceinline__ v8f wmma_f32(v2f a, v2f b, v8f c) {
  return __builtin_amdgcn_wmma_f32_16x16x4_f32(false, a, false, b, (short)0, c, false, false);
}
__device__ __forceinline__ v8f wmma_bf16(v16bf a, v16bf b, v8f c) {
  return __builtin_amdgcn_wmma_f32_16x16x32_bf16(false, a, false, b, (short)0, c, false, false);
}

// split fp32 into bf16 hi (truncated) + bf16 lo (residual)
__device__ __forceinline__ void split_bf16(float x, ushort_t& hi, ushort_t& lo) {
  union { float f; unsigned int u; } a, h, r;
  a.f = x;
  hi  = (ushort_t)(a.u >> 16);
  h.u = (unsigned int)hi << 16;
  r.f = x - h.f;                      // exact in fp32
  lo  = (ushort_t)(r.u >> 16);
}
// pack two splits into dwords (for ds_store_b64 staging)
__device__ __forceinline__ void split2_pack(float x0, float x1, uint_t& hp, uint_t& lp) {
  ushort_t h0, l0, h1, l1;
  split_bf16(x0, h0, l0);
  split_bf16(x1, h1, l1);
  hp = (uint_t)h0 | ((uint_t)h1 << 16);
  lp = (uint_t)l0 | ((uint_t)l1 << 16);
}

// ---------------------------------------------------------------------------
// P0: weight prep -> transposed bf16 hi/lo (coalesced writes, L2-hot reads)
// ---------------------------------------------------------------------------
__global__ void k_wprep_gate(const float* __restrict__ Wg, ushort_t* __restrict__ thi,
                             ushort_t* __restrict__ tlo) {
  size_t idx = (size_t)blockIdx.x * 256 + threadIdx.x;      // over 1024*2048
  int n = (int)(idx >> 11);
  int k = (int)(idx & (KGATE - 1));
  float x = Wg[(size_t)k * D_MODEL + n];
  ushort_t h, l; split_bf16(x, h, l);
  thi[idx] = h; tlo[idx] = l;
}
__global__ void k_wprep_plu(const float* __restrict__ Wp, ushort_t* __restrict__ thi,
                            ushort_t* __restrict__ tlo) {
  size_t idx = (size_t)blockIdx.x * 256 + threadIdx.x;      // over 1024*512
  int n = (int)(idx >> 9);
  int k = (int)(idx & (PSTRIDE - 1));
  float x = (k < PDIM) ? Wp[(size_t)k * D_MODEL + n] : 0.f;
  ushort_t h, l; split_bf16(x, h, l);
  thi[idx] = h; tlo[idx] = l;
}

// ---------------------------------------------------------------------------
// K1: z = h @ W_red + b_red  (fp32 WMMA: tiny GEMM, full precision for the
//     nonlinear Plucker stage)
// ---------------------------------------------------------------------------
__global__ void k_zred(const float* __restrict__ h, const float* __restrict__ Wred,
                       const float* __restrict__ bred, float* __restrict__ z) {
  const int wave  = threadIdx.x >> 5;
  const int lane  = threadIdx.x & 31;
  const int mtile = blockIdx.x * 8 + wave;
  const int m     = (mtile << 4) + (lane & 15);
  const int kb    = (lane >> 4) << 1;
  const int n0    = lane & 15;
  const float* hrow = h + (size_t)m * D_MODEL;

  v8f c0 = {}; v8f c1 = {};
  for (int k = 0; k < D_MODEL; k += 4) {
    v2f a = { hrow[k + kb], hrow[k + kb + 1] };
    const float* w0 = Wred + (size_t)(k + kb) * RDIM;
    const float* w1 = w0 + RDIM;
    v2f b0 = { w0[n0],      w1[n0]      };
    v2f b1 = { w0[n0 + 16], w1[n0 + 16] };
    c0 = wmma_f32(a, b0, c0);
    c1 = wmma_f32(a, b1, c1);
  }
  const int hi    = lane >> 4;
  const int mbase = (mtile << 4) + (hi << 3);
#pragma unroll
  for (int r = 0; r < 8; ++r) {
    size_t row = (size_t)(mbase + r) * RDIM;
    z[row + n0]      = c0[r] + bred[n0];
    z[row + 16 + n0] = c1[r] + bred[n0 + 16];
  }
}

// ---------------------------------------------------------------------------
// K2: sbar[t] = (1/max(count,1)) * sum_delta p_hat(t,delta)  (496-d, pad 512)
// ---------------------------------------------------------------------------
__global__ void k_pluck(const float* __restrict__ z, float* __restrict__ sbar) {
  const int wave = threadIdx.x >> 5;
  const int lane = threadIdx.x & 31;
  const int t    = blockIdx.x * 8 + wave;
  const int pos  = t & (SEQ - 1);

  int pi[16], pj[16];
#pragma unroll
  for (int j = 0; j < 16; ++j) {
    int k = lane + 32 * j;
    if (k < PDIM) {
      int i = 0, off = 0;
      while (off + (31 - i) <= k) { off += 31 - i; ++i; }
      pi[j] = i; pj[j] = i + 1 + (k - off);
    } else { pi[j] = 0; pj[j] = 0; }
  }

  const float z1 = z[(size_t)t * RDIM + lane];
  float acc[16];
#pragma unroll
  for (int j = 0; j < 16; ++j) acc[j] = 0.f;
  int count = 0;

  const int deltas[6] = {1, 2, 4, 8, 12, 16};
#pragma unroll
  for (int d = 0; d < 6; ++d) {
    const int delta = deltas[d];
    if (pos < delta) continue;
    ++count;
    const float z2 = z[(size_t)(t - delta) * RDIM + lane];
    float p[16];
    float ss = 0.f;
#pragma unroll
    for (int j = 0; j < 16; ++j) {
      float z1i = __shfl(z1, pi[j], 32);
      float z1j = __shfl(z1, pj[j], 32);
      float z2i = __shfl(z2, pi[j], 32);
      float z2j = __shfl(z2, pj[j], 32);
      p[j] = z1i * z2j - z1j * z2i;
      ss += p[j] * p[j];
    }
#pragma unroll
    for (int o = 16; o >= 1; o >>= 1) ss += __shfl_xor(ss, o, 32);
    float inv = 1.f / fmaxf(sqrtf(ss), 1e-8f);
#pragma unroll
    for (int j = 0; j < 16; ++j) acc[j] += p[j] * inv;
  }

  const float scale = 1.f / (float)(count > 0 ? count : 1);
  float* dst = sbar + (size_t)t * PSTRIDE;
#pragma unroll
  for (int j = 0; j < 16; ++j) dst[lane + 32 * j] = acc[j] * scale;
}

// ---------------------------------------------------------------------------
// K3: g = sbar @ W_plu + b_plu  via bf16x3 WMMA (K padded to 512)
// Block: 16 tokens x 1024 cols; 8 waves x 8 N-tiles.
// ---------------------------------------------------------------------------
#define AS3_STRIDE 520
__global__ void k_gplu(const float* __restrict__ sbar, const ushort_t* __restrict__ Bhi,
                       const ushort_t* __restrict__ Blo, const float* __restrict__ bplu,
                       float* __restrict__ g) {
  __shared__ __align__(16) ushort_t Ahi[16 * AS3_STRIDE];
  __shared__ __align__(16) ushort_t Alo[16 * AS3_STRIDE];
  const int tbase = blockIdx.x * 16;

  for (int i = threadIdx.x; i < 16 * PSTRIDE; i += 256) {
    int r = i >> 9, ci = i & (PSTRIDE - 1);
    ushort_t hh, ll;
    split_bf16(sbar[(size_t)tbase * PSTRIDE + i], hh, ll);
    Ahi[r * AS3_STRIDE + ci] = hh;
    Alo[r * AS3_STRIDE + ci] = ll;
  }
  __syncthreads();

  const int wave = threadIdx.x >> 5;
  const int lane = threadIdx.x & 31;
  const int m    = lane & 15;
  const int hi   = lane >> 4;
  const int n0   = lane & 15;

  v8f c[8];
#pragma unroll
  for (int nt = 0; nt < 8; ++nt) c[nt] = (v8f){};

  for (int k = 0; k < PSTRIDE; k += 32) {
    FragBF ah, al;
    const ushort_t* pa = &Ahi[m * AS3_STRIDE + k + 8 * hi];
    const ushort_t* pl = &Alo[m * AS3_STRIDE + k + 8 * hi];
    ah.q[0] = *(const uint4*)(pa);       ah.q[1] = *(const uint4*)(pa + 16);
    al.q[0] = *(const uint4*)(pl);       al.q[1] = *(const uint4*)(pl + 16);
#pragma unroll
    for (int nt = 0; nt < 8; ++nt) {
      int n = wave * 128 + nt * 16 + n0;
      FragBF bh, bl;
      const ushort_t* pb = Bhi + (size_t)n * PSTRIDE + k + 8 * hi;
      const ushort_t* pc = Blo + (size_t)n * PSTRIDE + k + 8 * hi;
      bh.q[0] = *(const uint4*)(pb);     bh.q[1] = *(const uint4*)(pb + 16);
      bl.q[0] = *(const uint4*)(pc);     bl.q[1] = *(const uint4*)(pc + 16);
      c[nt] = wmma_bf16(ah.v, bh.v, c[nt]);
      c[nt] = wmma_bf16(ah.v, bl.v, c[nt]);
      c[nt] = wmma_bf16(al.v, bh.v, c[nt]);
    }
  }

#pragma unroll
  for (int nt = 0; nt < 8; ++nt) {
    int col = wave * 128 + nt * 16 + n0;
    float bv = bplu[col];
#pragma unroll
    for (int r = 0; r < 8; ++r) {
      int token = tbase + r + (hi << 3);
      float val = c[nt][r] + bv;
      if ((token & (SEQ - 1)) == 0) val = 0.f;
      g[(size_t)token * D_MODEL + col] = val;
    }
  }
}

// ---------------------------------------------------------------------------
// K4: alpha = sigmoid([h,g] @ W_gate + b_gate) via bf16x3 WMMA.
// Block = 32 tokens x 512 cols (nhalf); 8 waves, each 2 M-tiles x 4 N-tiles
// (B fragments reused for 6 WMMAs). Double-buffered LDS A staging with
// next-chunk global loads issued before the barrier (latency overlap).
// Partial LN sums go to stats[t*4 + nhalf*2 + {0,1}] (deterministic merge).
// ---------------------------------------------------------------------------
#define AS4_STRIDE 136   // ushort stride: 272B row -> 4m mod 64 banks, 16B aligned
__global__ void k_gate(const float* __restrict__ h, const float* __restrict__ g,
                       const ushort_t* __restrict__ Bhi, const ushort_t* __restrict__ Blo,
                       const float* __restrict__ bg,
                       float* __restrict__ hmix, float* __restrict__ stats) {
  __shared__ __align__(16) ushort_t AhiB[2][32 * AS4_STRIDE];
  __shared__ __align__(16) ushort_t AloB[2][32 * AS4_STRIDE];
  __shared__ float redsum[32];
  __shared__ float redsq[32];

  const int tg    = blockIdx.x >> 1;
  const int nhalf = blockIdx.x & 1;
  const int tbase = tg * 32;
  const int wave  = threadIdx.x >> 5;
  const int lane  = threadIdx.x & 31;
  const int m     = lane & 15;
  const int hi    = lane >> 4;
  const int n0    = lane & 15;
  const int nbase = nhalf * 512 + wave * 64;          // wave covers 64 cols

  if (threadIdx.x < 32) { redsum[threadIdx.x] = 0.f; redsq[threadIdx.x] = 0.f; }

  v8f c[2][4];
#pragma unroll
  for (int mt = 0; mt < 2; ++mt)
#pragma unroll
    for (int nt = 0; nt < 4; ++nt) c[mt][nt] = (v8f){};

  // staging registers: rows r = wave + j*8, float4 col = lane
  float4 rv[4];
  {
    const float* src = h;
#pragma unroll
    for (int j = 0; j < 4; ++j) {
      int r = wave + j * 8;
      rv[j] = ((const float4*)(src + (size_t)(tbase + r) * D_MODEL))[lane];
    }
  }

  for (int ch = 0; ch < 16; ++ch) {                   // 16 chunks of K=128
    const int buf = ch & 1;
    // store staged chunk -> LDS (packed bf16 hi/lo, ds_store_b64)
#pragma unroll
    for (int j = 0; j < 4; ++j) {
      int r = wave + j * 8;
      uint_t h01, l01, h23, l23;
      split2_pack(rv[j].x, rv[j].y, h01, l01);
      split2_pack(rv[j].z, rv[j].w, h23, l23);
      uint2 hp = { h01, h23 };
      uint2 lp = { l01, l23 };
      *(uint2*)&AhiB[buf][r * AS4_STRIDE + lane * 4] = hp;
      *(uint2*)&AloB[buf][r * AS4_STRIDE + lane * 4] = lp;
    }
    // issue next chunk's global loads before the barrier (overlap with WMMA)
    if (ch < 15) {
      const int   nc  = ch + 1;
      const float* src = (nc < 8) ? h : g;
      const int   kofs4 = ((nc & 7) * 128) >> 2;
#pragma unroll
      for (int j = 0; j < 4; ++j) {
        int r = wave + j * 8;
        rv[j] = ((const float4*)(src + (size_t)(tbase + r) * D_MODEL))[kofs4 + lane];
      }
    }
    __syncthreads();

    const int kbase = ((ch < 8) ? 0 : D_MODEL) + (ch & 7) * 128;
    for (int kl = 0; kl < 128; kl += 32) {            // 4 K-slabs per chunk
      FragBF ah0, al0, ah1, al1;
      {
        const ushort_t* p0h = &AhiB[buf][(m)      * AS4_STRIDE + kl + 8 * hi];
        const ushort_t* p0l = &AloB[buf][(m)      * AS4_STRIDE + kl + 8 * hi];
        const ushort_t* p1h = &AhiB[buf][(16 + m) * AS4_STRIDE + kl + 8 * hi];
        const ushort_t* p1l = &AloB[buf][(16 + m) * AS4_STRIDE + kl + 8 * hi];
        ah0.q[0] = *(const uint4*)(p0h); ah0.q[1] = *(const uint4*)(p0h + 16);
        al0.q[0] = *(const uint4*)(p0l); al0.q[1] = *(const uint4*)(p0l + 16);
        ah1.q[0] = *(const uint4*)(p1h); ah1.q[1] = *(const uint4*)(p1h + 16);
        al1.q[0] = *(const uint4*)(p1l); al1.q[1] = *(const uint4*)(p1l + 16);
      }
      const int kg = kbase + kl;
#pragma unroll
      for (int nt = 0; nt < 4; ++nt) {
        int n = nbase + nt * 16 + n0;
        FragBF bh, bl;
        const ushort_t* pb = Bhi + (size_t)n * KGATE + kg + 8 * hi;
        const ushort_t* pc = Blo + (size_t)n * KGATE + kg + 8 * hi;
        bh.q[0] = *(const uint4*)(pb);   bh.q[1] = *(const uint4*)(pb + 16);
        bl.q[0] = *(const uint4*)(pc);   bl.q[1] = *(const uint4*)(pc + 16);
        c[0][nt] = wmma_bf16(ah0.v, bh.v, c[0][nt]);
        c[0][nt] = wmma_bf16(ah0.v, bl.v, c[0][nt]);
        c[0][nt] = wmma_bf16(al0.v, bh.v, c[0][nt]);
        c[1][nt] = wmma_bf16(ah1.v, bh.v, c[1][nt]);
        c[1][nt] = wmma_bf16(ah1.v, bl.v, c[1][nt]);
        c[1][nt] = wmma_bf16(al1.v, bh.v, c[1][nt]);
      }
    }
  }

  float psum[2][8], psq[2][8];
#pragma unroll
  for (int mt = 0; mt < 2; ++mt)
#pragma unroll
    for (int r = 0; r < 8; ++r) { psum[mt][r] = 0.f; psq[mt][r] = 0.f; }

#pragma unroll
  for (int nt = 0; nt < 4; ++nt) {
    int col   = nbase + nt * 16 + n0;
    float bgv = bg[col];
#pragma unroll
    for (int mt = 0; mt < 2; ++mt) {
#pragma unroll
      for (int r = 0; r < 8; ++r) {
        int row    = mt * 16 + r + (hi << 3);
        size_t idx = (size_t)(tbase + row) * D_MODEL + col;
        float a  = 1.f / (1.f + __expf(-(c[mt][nt][r] + bgv)));
        float hv = h[idx];
        float gv = g[idx];
        float mx = a * hv + (1.f - a) * gv;
        hmix[idx] = mx;
        psum[mt][r] += mx;
        psq[mt][r]  += mx * mx;
      }
    }
  }
#pragma unroll
  for (int mt = 0; mt < 2; ++mt)
#pragma unroll
    for (int r = 0; r < 8; ++r) {
#pragma unroll
      for (int o = 8; o >= 1; o >>= 1) {
        psum[mt][r] += __shfl_xor(psum[mt][r], o, 32);
        psq[mt][r]  += __shfl_xor(psq[mt][r],  o, 32);
      }
    }
  __syncthreads();
  if (n0 == 0) {
#pragma unroll
    for (int mt = 0; mt < 2; ++mt)
#pragma unroll
      for (int r = 0; r < 8; ++r) {
        atomicAdd(&redsum[mt * 16 + r + (hi << 3)], psum[mt][r]);   // ds_add_f32
        atomicAdd(&redsq [mt * 16 + r + (hi << 3)], psq[mt][r]);
      }
  }
  __syncthreads();
  if (threadIdx.x < 32) {
    size_t t = (size_t)(tbase + threadIdx.x);
    stats[t * 4 + nhalf * 2]     = redsum[threadIdx.x];
    stats[t * 4 + nhalf * 2 + 1] = redsq[threadIdx.x];
  }
}

// ---------------------------------------------------------------------------
// K5: in-place LayerNorm of d_out; merges the two column-half partials.
// ---------------------------------------------------------------------------
__global__ void k_ln(const float* __restrict__ stats, const float* __restrict__ lng,
                     const float* __restrict__ lnb, float* __restrict__ out) {
  const int t  = blockIdx.x;
  const float s   = stats[(size_t)t * 4]     + stats[(size_t)t * 4 + 2];
  const float sq  = stats[(size_t)t * 4 + 1] + stats[(size_t)t * 4 + 3];
  const float mu  = s * (1.f / (float)D_MODEL);
  const float var = sq * (1.f / (float)D_MODEL) - mu * mu;
  const float inv = rsqrtf(var + 1e-5f);
  for (int cI = threadIdx.x; cI < D_MODEL; cI += 256) {
    size_t idx = (size_t)t * D_MODEL + cI;
    out[idx] = (out[idx] - mu) * inv * lng[cI] + lnb[cI];
  }
}

// ---------------------------------------------------------------------------
extern "C" void kernel_launch(void* const* d_in, const int* in_sizes, int n_in,
                              void* d_out, int out_size, void* d_ws, size_t ws_size,
                              hipStream_t stream) {
  const float* h     = (const float*)d_in[0];
  const float* W_red = (const float*)d_in[1];
  const float* b_red = (const float*)d_in[2];
  const float* W_plu = (const float*)d_in[3];
  const float* b_plu = (const float*)d_in[4];
  const float* W_gat = (const float*)d_in[5];
  const float* b_gat = (const float*)d_in[6];
  const float* ln_g  = (const float*)d_in[7];
  const float* ln_b  = (const float*)d_in[8];
  float* out = (float*)d_out;
  float* ws  = (float*)d_ws;

  float*    z_buf = ws + Z_OFF;
  float*    sbar  = ws + SBAR_OFF;
  float*    g_buf = ws + G_OFF;
  float*    stats = ws + STATS_OFF;
  ushort_t* ws_u  = (ushort_t*)(ws + FLOAT_END);
  ushort_t* wgHi  = ws_u + WGT_HI_OFF;
  ushort_t* wgLo  = ws_u + WGT_LO_OFF;
  ushort_t* wpHi  = ws_u + WPT_HI_OFF;
  ushort_t* wpLo  = ws_u + WPT_LO_OFF;

  k_wprep_gate<<<(D_MODEL * KGATE) / 256,   256, 0, stream>>>(W_gat, wgHi, wgLo);
  k_wprep_plu <<<(D_MODEL * PSTRIDE) / 256, 256, 0, stream>>>(W_plu, wpHi, wpLo);
  k_zred <<<NTOK / (16 * 8), 256, 0, stream>>>(h, W_red, b_red, z_buf);
  k_pluck<<<NTOK / 8,        256, 0, stream>>>(z_buf, sbar);
  k_gplu <<<NTOK / 16,       256, 0, stream>>>(sbar, wpHi, wpLo, b_plu, g_buf);
  k_gate <<<(NTOK / 32) * 2, 256, 0, stream>>>(h, g_buf, wgHi, wgLo, b_gat, out, stats);
  k_ln   <<<NTOK,            256, 0, stream>>>(stats, ln_g, ln_b, out);
}
